// SSM_spa_56427280335363
// MI455X (gfx1250) — compile-verified, matrix operand order checked by reading, not run on hardware
//
#include <hip/hip_runtime.h>
#include <stdint.h>

// ---------------------------------------------------------------------------
// Types for CDNA5 WMMA (wave32, 16x16x32 bf16 -> f32)
// ---------------------------------------------------------------------------
typedef __attribute__((ext_vector_type(16))) __bf16 v16bf;
typedef __attribute__((ext_vector_type(8)))  __bf16 v8bf;
typedef __attribute__((ext_vector_type(8)))  float  v8f;

// Problem constants (match reference)
#define BATCH    8
#define L_SEQ    256      // sequence length == C_IN
#define DMODEL   1024
#define DINNER   2048
#define DSTATE   16
#define DTRANK   64
#define NPROJ    96       // DT_RANK + 2*D_STATE (logical)
#define NPROJ_P  128      // padded to a multiple of 64 for the GEMM fast path
#define HH       64
#define WW       64
#define SS       32       // SIZE
#define ROWS     (BATCH * L_SEQ)   // 2048 rows for all sequence GEMMs

// ---------------------------------------------------------------------------
// helpers
// ---------------------------------------------------------------------------
static __device__ __forceinline__ unsigned short f2bf(float f) {
  unsigned int u = __float_as_uint(f);
  unsigned int r = (u + 0x7FFFu + ((u >> 16) & 1u)) >> 16;  // RNE
  return (unsigned short)r;
}
static __device__ __forceinline__ float silu(float x) {
  return x / (1.0f + __expf(-x));
}
static __device__ __forceinline__ float softplus(float x) {
  return (x > 20.0f) ? x : log1pf(__expf(x));
}

// ---------------------------------------------------------------------------
// fp32 -> bf16 bit convert (weights, activations)
// ---------------------------------------------------------------------------
__global__ void k_f32_to_bf16(const float* __restrict__ in,
                              unsigned short* __restrict__ out, int n) {
  int i = blockIdx.x * blockDim.x + threadIdx.x;
  if (i < n) out[i] = f2bf(in[i]);
}

// Pad x_proj_w [2048,96] -> bf16 [2048,128] with zero columns 96..127.
__global__ void k_pad_xproj(const float* __restrict__ in,
                            unsigned short* __restrict__ out) {
  int i = blockIdx.x * blockDim.x + threadIdx.x;
  if (i >= DINNER * NPROJ_P) return;
  int r = i >> 7, n = i & (NPROJ_P - 1);
  out[i] = (n < NPROJ) ? f2bf(in[r * NPROJ + n]) : (unsigned short)0;
}

// ---------------------------------------------------------------------------
// Bilinear downsample 64x64 -> 32x32 (align_corners), write bf16 rows
// ---------------------------------------------------------------------------
__global__ void k_resize_down(const float* __restrict__ x,
                              unsigned short* __restrict__ xs) {
  int i = blockIdx.x * blockDim.x + threadIdx.x;
  if (i >= BATCH * L_SEQ * SS * SS) return;
  int sx = i & (SS - 1);
  int sy = (i >> 5) & (SS - 1);
  int bc = i >> 10;                       // b*256 + c
  const float sc = (float)(HH - 1) / (float)(SS - 1);
  float py = sy * sc, px = sx * sc;
  int y0 = (int)py, x0 = (int)px;
  int y1 = min(y0 + 1, HH - 1), x1 = min(x0 + 1, WW - 1);
  float ty = py - (float)y0, tx = px - (float)x0;
  const float* img = x + (size_t)bc * (HH * WW);
  float v00 = img[y0 * WW + x0], v01 = img[y0 * WW + x1];
  float v10 = img[y1 * WW + x0], v11 = img[y1 * WW + x1];
  float v = (1.f - ty) * ((1.f - tx) * v00 + tx * v01)
          + ty * ((1.f - tx) * v10 + tx * v11);
  xs[(size_t)bc * (SS * SS) + sy * SS + sx] = f2bf(v);
}

// ---------------------------------------------------------------------------
// Tiled WMMA GEMM (fast path): C[M,N] = A[M,K](bf16,rm) * B[K,N](bf16,rm)+bias
// Requirements: M%64==0, N%64==0, K%32==0 (all call sites satisfy this).
// Block: 256 threads = 8 waves; tile 64(M) x 64(N); K-step 32.
// Wave grid 4x2: each wave owns a 16x32 strip -> 2 WMMAs/K-step, A reused.
// ---------------------------------------------------------------------------
#define TM 64
#define TN 64
#define TK 32
#define LDT 40  // padded LDS row (halves); 80B rows keep 16B alignment

__global__ __launch_bounds__(256) void k_gemm_bf16(
    const unsigned short* __restrict__ A, const unsigned short* __restrict__ B,
    float* __restrict__ C, const float* __restrict__ bias,
    int M, int N, int K) {
  __shared__ __align__(16) unsigned short As[TM][LDT];
  __shared__ __align__(16) unsigned short Bs[TN][LDT];

  const int tid   = threadIdx.x;
  const int lane  = tid & 31;
  const int wave  = tid >> 5;      // 0..7
  const int wm    = wave >> 1;     // 0..3  (M sub-tile, 16 rows)
  const int wn    = wave & 1;      // 0..1  (N sub-tile, 32 cols)
  const int m0    = blockIdx.y * TM;
  const int n0    = blockIdx.x * TN;
  const int frow  = lane & 15;     // fragment row (M for A, N for B/C)
  const int khalf = lane >> 4;     // which K-half this lane owns

  // staging coordinates (uniform per thread across K-steps)
  const int ar = tid >> 2;                 // 0..63  A row
  const int ak = (tid & 3) * 8;            // 0,8,16,24 (8 halves = 16B)
  const int bk = tid >> 3;                 // 0..31  B k-row
  const int bn = (tid & 7) * 8;            // 0..56  (8 halves = 16B)

  const unsigned short* Aptr = A + (size_t)(m0 + ar) * K + ak;
  const unsigned short* Bptr = B + (size_t)bk * N + n0 + bn;

  v8f acc0 = {};
  v8f acc1 = {};

  for (int k0 = 0; k0 < K; k0 += TK) {
    // --- stage A tile (64x32): one b128 per thread, coalesced ---
    *(uint4*)&As[ar][ak] = *(const uint4*)(Aptr + k0);
    // --- stage B tile (32x64) transposed to Bs[n][k]: b128 + 8 ds_store_b16
    {
      union { uint4 u; unsigned short s[8]; } t;
      t.u = *(const uint4*)(Bptr + (size_t)k0 * N);
      #pragma unroll
      for (int j = 0; j < 8; ++j) Bs[bn + j][bk] = t.s[j];
    }
    // prefetch next K tiles while this one is consumed
    if (k0 + TK < K) {
      __builtin_prefetch(Aptr + k0 + TK, 0, 1);
      __builtin_prefetch(Bptr + (size_t)(k0 + TK) * N, 0, 1);
    }
    __syncthreads();

    // --- per-lane fragments (ISA 16-bit operand layout), ds_load_b128 x2 ---
    v16bf af, bf0, bf1;
    ((v8bf*)&af)[0]  = *(const v8bf*)&As[wm * 16 + frow][khalf * 8];
    ((v8bf*)&af)[1]  = *(const v8bf*)&As[wm * 16 + frow][16 + khalf * 8];
    ((v8bf*)&bf0)[0] = *(const v8bf*)&Bs[wn * 32 + frow][khalf * 8];
    ((v8bf*)&bf0)[1] = *(const v8bf*)&Bs[wn * 32 + frow][16 + khalf * 8];
    ((v8bf*)&bf1)[0] = *(const v8bf*)&Bs[wn * 32 + 16 + frow][khalf * 8];
    ((v8bf*)&bf1)[1] = *(const v8bf*)&Bs[wn * 32 + 16 + frow][16 + khalf * 8];

    acc0 = __builtin_amdgcn_wmma_f32_16x16x32_bf16(
        false, af, false, bf0, (short)0, acc0, false, false);
    acc1 = __builtin_amdgcn_wmma_f32_16x16x32_bf16(
        false, af, false, bf1, (short)0, acc1, false, false);
    __syncthreads();
  }

  // --- store: C/D layout: lane -> N = lane&15, VGPR r -> M = r + (lane>>4)*8
  const int n = n0 + wn * 32 + frow;
  const float bv0 = bias ? bias[n] : 0.0f;
  const float bv1 = bias ? bias[n + 16] : 0.0f;
  const int mbase = m0 + wm * 16 + khalf * 8;
  #pragma unroll
  for (int r = 0; r < 8; ++r) {
    C[(size_t)(mbase + r) * N + n]      = acc0[r] + bv0;
    C[(size_t)(mbase + r) * N + n + 16] = acc1[r] + bv1;
  }
}

// ---------------------------------------------------------------------------
// Causal depthwise conv1d along L (kernel 4, left pad 3) + SiLU.
// Reads xi = xz[:, :2048]; writes xc fp32 and bf16.
// ---------------------------------------------------------------------------
__global__ void k_conv_silu(const float* __restrict__ xz,
                            const float* __restrict__ conv_w,
                            const float* __restrict__ conv_b,
                            float* __restrict__ xc_f,
                            unsigned short* __restrict__ xc_bf) {
  int i = blockIdx.x * blockDim.x + threadIdx.x;
  if (i >= BATCH * L_SEQ * DINNER) return;
  int d = i & (DINNER - 1);
  int l = (i >> 11) & (L_SEQ - 1);
  int b = i >> 19;
  float s = conv_b[d];
  #pragma unroll
  for (int k = 0; k < 4; ++k) {
    int ls = l + k - 3;
    if (ls >= 0)
      s += xz[((size_t)(b * L_SEQ + ls)) * (2 * DINNER) + d] * conv_w[d * 4 + k];
  }
  s = silu(s);
  size_t o = (size_t)(b * L_SEQ + l) * DINNER + d;
  xc_f[o]  = s;
  xc_bf[o] = f2bf(s);
}

// ---------------------------------------------------------------------------
// Extract dt-rank slice of padded proj and convert to bf16.
// ---------------------------------------------------------------------------
__global__ void k_extract_dt(const float* __restrict__ proj,
                             unsigned short* __restrict__ dtA) {
  int i = blockIdx.x * blockDim.x + threadIdx.x;
  if (i >= ROWS * DTRANK) return;
  int r = i >> 6, j = i & 63;
  dtA[i] = f2bf(proj[(size_t)r * NPROJ_P + j]);
}

// ---------------------------------------------------------------------------
// Selective scan (S6). One thread per (b,d); D_STATE=16 states in registers.
// Per-step B/C vectors staged in LDS. Epilogue: +x*D, *silu(z), bf16 out.
// ---------------------------------------------------------------------------
__global__ __launch_bounds__(256) void k_scan(
    const float* __restrict__ proj,     // [ROWS, 128] (B at +64, C at +80)
    const float* __restrict__ dt_raw,   // [ROWS, DINNER] pre-softplus
    const float* __restrict__ xc,       // [ROWS, DINNER]
    const float* __restrict__ xz,       // [ROWS, 2*DINNER] (z at +DINNER)
    const float* __restrict__ A_log,    // [DINNER, 16]
    const float* __restrict__ Dp,       // [DINNER]
    unsigned short* __restrict__ ybf) { // [ROWS, DINNER]
  int b = blockIdx.x >> 3;
  int d = ((blockIdx.x & 7) << 8) + threadIdx.x;
  float Ar[DSTATE], h[DSTATE];
  #pragma unroll
  for (int s = 0; s < DSTATE; ++s) {
    Ar[s] = -__expf(A_log[(size_t)d * DSTATE + s]);
    h[s] = 0.0f;
  }
  float Dd = Dp[d];
  __shared__ float BC[2 * DSTATE];
  for (int l = 0; l < L_SEQ; ++l) {
    int row = b * L_SEQ + l;
    __syncthreads();
    if (threadIdx.x < 2 * DSTATE)
      BC[threadIdx.x] = proj[(size_t)row * NPROJ_P + DTRANK + threadIdx.x];
    __syncthreads();
    float dt  = softplus(dt_raw[(size_t)row * DINNER + d]);
    float xv  = xc[(size_t)row * DINNER + d];
    float dtx = dt * xv;
    float y = 0.0f;
    #pragma unroll
    for (int s = 0; s < DSTATE; ++s) {
      float dA = __expf(dt * Ar[s]);
      h[s] = dA * h[s] + dtx * BC[s];
      y += h[s] * BC[DSTATE + s];
    }
    float zv = xz[(size_t)row * (2 * DINNER) + DINNER + d];
    y = (y + xv * Dd) * silu(zv);
    ybf[(size_t)row * DINNER + d] = f2bf(y);
  }
}

// ---------------------------------------------------------------------------
// LayerNorm over last dim (1024) per row; write transposed bf16 [b, s, c]
// so the channel-mix GEMM reads a row-major A.
// ---------------------------------------------------------------------------
__global__ __launch_bounds__(256) void k_layernorm_T(
    const float* __restrict__ y,        // [ROWS, DMODEL]
    const float* __restrict__ g, const float* __restrict__ be,
    unsigned short* __restrict__ yt) {  // [BATCH*DMODEL, L_SEQ]
  __shared__ float red[256];
  int r = blockIdx.x;            // 0..2047  (b*256 + c)
  int b = r >> 8, c = r & 255;
  const float* yr = y + (size_t)r * DMODEL;
  int tid = threadIdx.x;
  float lsum = 0.f, lsq = 0.f;
  for (int j = tid; j < DMODEL; j += 256) {
    float v = yr[j];
    lsum += v; lsq += v * v;
  }
  red[tid] = lsum; __syncthreads();
  for (int s = 128; s > 0; s >>= 1) {
    if (tid < s) red[tid] += red[tid + s];
    __syncthreads();
  }
  float mean = red[0] * (1.0f / DMODEL);
  __syncthreads();
  red[tid] = lsq; __syncthreads();
  for (int s = 128; s > 0; s >>= 1) {
    if (tid < s) red[tid] += red[tid + s];
    __syncthreads();
  }
  float var = red[0] * (1.0f / DMODEL) - mean * mean;
  float rs = rsqrtf(var + 1e-5f);
  for (int j = tid; j < DMODEL; j += 256) {
    float v = (yr[j] - mean) * rs * g[j] + be[j];
    yt[((size_t)b * DMODEL + j) * L_SEQ + c] = f2bf(v);
  }
}

// ---------------------------------------------------------------------------
// Bilinear upsample 32x32 -> 64x64 (align_corners) of channel-mixed result.
// small layout: [b, s=sy*32+sx, o]; out layout: [b, o, oy, ox] fp32.
// ---------------------------------------------------------------------------
__global__ void k_resize_up(const float* __restrict__ small,
                            float* __restrict__ out) {
  int i = blockIdx.x * blockDim.x + threadIdx.x;
  if (i >= BATCH * 256 * HH * WW) return;
  int ox = i & (WW - 1);
  int oy = (i >> 6) & (HH - 1);
  int o  = (i >> 12) & 255;
  int b  = i >> 20;
  const float sc = (float)(SS - 1) / (float)(HH - 1);
  float py = oy * sc, px = ox * sc;
  int y0 = (int)py, x0 = (int)px;
  int y1 = min(y0 + 1, SS - 1), x1 = min(x0 + 1, SS - 1);
  float ty = py - (float)y0, tx = px - (float)x0;
  const float* base = small + (size_t)b * (SS * SS * 256) + o;
  float v00 = base[(y0 * SS + x0) * 256], v01 = base[(y0 * SS + x1) * 256];
  float v10 = base[(y1 * SS + x0) * 256], v11 = base[(y1 * SS + x1) * 256];
  out[i] = (1.f - ty) * ((1.f - tx) * v00 + tx * v01)
         + ty * ((1.f - tx) * v10 + tx * v11);
}

// ---------------------------------------------------------------------------
// Launcher
// ---------------------------------------------------------------------------
static inline int cdiv(int a, int b) { return (a + b - 1) / b; }

extern "C" void kernel_launch(void* const* d_in, const int* in_sizes, int n_in,
                              void* d_out, int out_size, void* d_ws, size_t ws_size,
                              hipStream_t stream) {
  const float* x1       = (const float*)d_in[0];
  const float* in_proj  = (const float*)d_in[1];
  const float* conv_w   = (const float*)d_in[2];
  const float* conv_b   = (const float*)d_in[3];
  const float* x_proj   = (const float*)d_in[4];
  const float* dt_proj  = (const float*)d_in[5];
  const float* dt_b     = (const float*)d_in[6];
  const float* A_log    = (const float*)d_in[7];
  const float* D_param  = (const float*)d_in[8];
  const float* out_proj = (const float*)d_in[9];
  const float* ln_g     = (const float*)d_in[10];
  const float* ln_b     = (const float*)d_in[11];
  const float* mix_w    = (const float*)d_in[12];
  const float* mix_b    = (const float*)d_in[13];
  float* out = (float*)d_out;

  // ---- carve workspace ----
  char* ws = (char*)d_ws;
  size_t off = 0;
  auto carve = [&](size_t bytes) -> void* {
    off = (off + 255) & ~(size_t)255;
    void* p = ws + off;
    off += bytes;
    return p;
  };
  unsigned short* xs_bf  = (unsigned short*)carve((size_t)ROWS * DMODEL * 2);
  unsigned short* w_in   = (unsigned short*)carve((size_t)DMODEL * 2 * DINNER * 2);
  unsigned short* w_xp   = (unsigned short*)carve((size_t)DINNER * NPROJ_P * 2);
  unsigned short* w_dt   = (unsigned short*)carve((size_t)DTRANK * DINNER * 2);
  unsigned short* w_out  = (unsigned short*)carve((size_t)DINNER * DMODEL * 2);
  unsigned short* w_mix  = (unsigned short*)carve((size_t)256 * 256 * 2);
  float*          xz     = (float*)carve((size_t)ROWS * 2 * DINNER * 4);
  float*          xc_f   = (float*)carve((size_t)ROWS * DINNER * 4);
  unsigned short* xc_bf  = (unsigned short*)carve((size_t)ROWS * DINNER * 2);
  float*          proj   = (float*)carve((size_t)ROWS * NPROJ_P * 4);
  unsigned short* dtA    = (unsigned short*)carve((size_t)ROWS * DTRANK * 2);
  float*          dt_raw = (float*)carve((size_t)ROWS * DINNER * 4);
  unsigned short* y_bf   = (unsigned short*)carve((size_t)ROWS * DINNER * 2);
  float*          ymm    = (float*)carve((size_t)ROWS * DMODEL * 4);
  unsigned short* yt_bf  = (unsigned short*)carve((size_t)BATCH * DMODEL * L_SEQ * 2);
  float*          small  = (float*)carve((size_t)BATCH * DMODEL * 256 * 4);
  (void)ws_size; (void)in_sizes; (void)n_in; (void)out_size;

  // ---- weight conversions (fp32 -> bf16 bits) ----
  {
    int n;
    n = DMODEL * 2 * DINNER;
    k_f32_to_bf16<<<cdiv(n, 256), 256, 0, stream>>>(in_proj, w_in, n);
    n = DINNER * NPROJ_P;
    k_pad_xproj<<<cdiv(n, 256), 256, 0, stream>>>(x_proj, w_xp);
    n = DTRANK * DINNER;
    k_f32_to_bf16<<<cdiv(n, 256), 256, 0, stream>>>(dt_proj, w_dt, n);
    n = DINNER * DMODEL;
    k_f32_to_bf16<<<cdiv(n, 256), 256, 0, stream>>>(out_proj, w_out, n);
    n = 256 * 256;
    k_f32_to_bf16<<<cdiv(n, 256), 256, 0, stream>>>(mix_w, w_mix, n);
  }

  // ---- downsample -> bf16 sequence rows [2048, 1024] ----
  {
    int n = BATCH * L_SEQ * SS * SS;
    k_resize_down<<<cdiv(n, 256), 256, 0, stream>>>(x1, xs_bf);
  }

  // ---- in_proj: [2048,1024] x [1024,4096] -> xz ----
  {
    dim3 g((2 * DINNER) / TN, ROWS / TM);
    k_gemm_bf16<<<g, 256, 0, stream>>>(xs_bf, w_in, xz, nullptr,
                                       ROWS, 2 * DINNER, DMODEL);
  }

  // ---- causal conv + SiLU ----
  {
    int n = BATCH * L_SEQ * DINNER;
    k_conv_silu<<<cdiv(n, 256), 256, 0, stream>>>(xz, conv_w, conv_b, xc_f, xc_bf);
  }

  // ---- x_proj (padded): [2048,2048] x [2048,128] -> proj ----
  {
    dim3 g(NPROJ_P / TN, ROWS / TM);
    k_gemm_bf16<<<g, 256, 0, stream>>>(xc_bf, w_xp, proj, nullptr,
                                       ROWS, NPROJ_P, DINNER);
  }

  // ---- dt GEMM: [2048,64] x [64,2048] + dt_b -> dt_raw ----
  {
    int n = ROWS * DTRANK;
    k_extract_dt<<<cdiv(n, 256), 256, 0, stream>>>(proj, dtA);
    dim3 g(DINNER / TN, ROWS / TM);
    k_gemm_bf16<<<g, 256, 0, stream>>>(dtA, w_dt, dt_raw, dt_b,
                                       ROWS, DINNER, DTRANK);
  }

  // ---- selective scan ----
  k_scan<<<BATCH * (DINNER / 256), 256, 0, stream>>>(
      proj, dt_raw, xc_f, xz, A_log, D_param, y_bf);

  // ---- out_proj: [2048,2048] x [2048,1024] -> ymm ----
  {
    dim3 g(DMODEL / TN, ROWS / TM);
    k_gemm_bf16<<<g, 256, 0, stream>>>(y_bf, w_out, ymm, nullptr,
                                       ROWS, DMODEL, DINNER);
  }

  // ---- LayerNorm + transpose -> yt_bf [B*1024, 256] ----
  k_layernorm_T<<<ROWS, 256, 0, stream>>>(ymm, ln_g, ln_b, yt_bf);

  // ---- channel mix at 32x32 (commutes with upsample):
  //      [8192,256] x [256,256] + bias -> small ----
  {
    dim3 g(256 / TN, (BATCH * DMODEL) / TM);
    k_gemm_bf16<<<g, 256, 0, stream>>>(yt_bf, w_mix, small, mix_b,
                                       BATCH * DMODEL, 256, 256);
  }

  // ---- bilinear upsample -> final fp32 output [8,256,64,64] ----
  {
    int n = BATCH * 256 * HH * WW;
    k_resize_up<<<cdiv(n, 256), 256, 0, stream>>>(small, out);
  }
}
